// Pointer_85315230367921
// MI455X (gfx1250) — matrix operand order repaired and checked by reading, not compiled, not run on hardware
//
#include <hip/hip_runtime.h>
#include <hip/hip_bf16.h>

// ---------------------------------------------------------------------------
// Problem constants (from reference): S=50, B=2048, H=128, D=2H=256, 49 steps
// ---------------------------------------------------------------------------
#define SEQ   50
#define BATCH 2048
#define HID   128
#define DIM   256
#define STEPS 49

typedef __bf16 bf16_t;
typedef __attribute__((ext_vector_type(16))) __bf16 v16bf;
typedef __attribute__((ext_vector_type(8)))  float  v8f;
typedef __attribute__((ext_vector_type(4))) unsigned int v4u_t;
typedef __attribute__((ext_vector_type(8))) int v8i_t;
typedef __attribute__((ext_vector_type(4))) int v4i_t;

union FragAB { uint32_t u[8]; v16bf v; };

__device__ __forceinline__ float sigmoidf_(float x) {
    return 1.0f / (1.0f + __expf(-x));
}

// ---------------------------------------------------------------------------
// Tensor Data Mover staging (gfx1250): one wave issues a 2D tile DMA
// (SEQ rows x DIM bf16, row stride BATCH*DIM elems) into LDS, waits
// TENSORcnt, then the workgroup barrier publishes the tile.
// This toolchain's builtin takes 6 args: (g0 u32x4, g1 i32x8, g2 i32x4,
// g3 i32x4, i32x8, i32 cpol) — extra group zeroed for a 2D tile.
// ---------------------------------------------------------------------------
#if defined(__gfx1250__) && __has_builtin(__builtin_amdgcn_tensor_load_to_lds) && \
    __has_builtin(__builtin_amdgcn_s_wait_tensorcnt)
#define USE_TDM 1
#else
#define USE_TDM 0
#endif

__device__ __forceinline__ unsigned lds_offset_of(const void* p) {
    // generic pointer to LDS: low 32 bits are the LDS byte address
    return (unsigned)(uintptr_t)p;
}

#if USE_TDM
__device__ __forceinline__ void tdm_load_2d(unsigned lds_addr, const void* gptr,
                                            unsigned tile_d0, unsigned tile_d1,
                                            unsigned tens_d0, unsigned tens_d1,
                                            unsigned long long d0_stride_elems)
{
    const unsigned long long ga = (unsigned long long)(uintptr_t)gptr;
    v4u_t g0;
    g0[0] = 1u;                                         // count=1, user descriptor
    g0[1] = lds_addr;                                   // LDS byte address
    g0[2] = (unsigned)(ga & 0xFFFFFFFFu);               // global addr [31:0]
    g0[3] = (unsigned)((ga >> 32) & 0x1FFFFFFu) | (2u << 30);  // addr[56:32], type=2
    v8i_t g1;
    g1[0] = (int)(1u << 16);                            // data_size=1 (2B), mask=0
    g1[1] = (int)((tens_d0 & 0xFFFFu) << 16);           // tensor_dim0[15:0]
    g1[2] = (int)(((tens_d0 >> 16) & 0xFFFFu) | ((tens_d1 & 0xFFFFu) << 16));
    g1[3] = (int)(((tens_d1 >> 16) & 0xFFFFu) | ((tile_d0 & 0xFFFFu) << 16));
    g1[4] = (int)(tile_d1 & 0xFFFFu);                   // tile_dim1, tile_dim2=0
    g1[5] = (int)(d0_stride_elems & 0xFFFFFFFFull);     // dim0 stride [31:0]
    g1[6] = (int)((d0_stride_elems >> 32) & 0xFFFFull); // stride[47:32], dim1_stride=0
    g1[7] = 0;
    const v4i_t z4 = {0, 0, 0, 0};
    const v8i_t z8 = {0, 0, 0, 0, 0, 0, 0, 0};
    __builtin_amdgcn_tensor_load_to_lds(g0, g1, z4, z4, z8, 0);
}
#endif

// Stage one SEQ x DIM bf16 tile (column block of a [S,B,D] tensor at batch b).
__device__ __forceinline__ void stage_tile(bf16_t* dstLds, const bf16_t* src,
                                           int b, int t)
{
#if USE_TDM
    if (t < 32) {
        tdm_load_2d(lds_offset_of(dstLds), src + (size_t)b * DIM,
                    DIM, SEQ, DIM, SEQ, (unsigned long long)BATCH * DIM);
    }
#else
    for (int idx = t; idx < SEQ * (DIM / 8); idx += 256) {
        const int s = idx >> 5;
        const int x = (idx & 31) * 8;
        *reinterpret_cast<uint4*>(dstLds + s * DIM + x) =
            *reinterpret_cast<const uint4*>(src + ((size_t)s * BATCH + b) * DIM + x);
    }
#endif
}

__device__ __forceinline__ void stage_wait(int t)
{
#if USE_TDM
    if (t < 32) __builtin_amdgcn_s_wait_tensorcnt(0);
#endif
    __syncthreads();
}

// ---------------------------------------------------------------------------
// bf16 WMMA GEMM:  C[M,N] = A[M,K] * W[N,K]^T (+ bias[N])
// Block = 256 threads = 8 waves; block tile 32(M) x 256(N); each wave owns a
// 16x64 strip (4 accumulators). Software-pipelined: next k-tile's fragments
// (10 x b128) are issued before the current tile's 4 WMMAs so global latency
// overlaps matrix math instead of s_wait_loadcnt 0 before every WMMA.
// Fragment addressing follows the CDNA5 16-bit A(16x32)/B(32x16) lane tables;
// each fragment is two contiguous 16-byte chunks -> global_load_b128 pairs.
// ---------------------------------------------------------------------------
template <bool OUT_BF16, bool HAS_BIAS>
__global__ __launch_bounds__(256)
void gemm_bf16_kernel(const bf16_t* __restrict__ A,
                      const bf16_t* __restrict__ W,
                      const float*  __restrict__ bias,
                      void* __restrict__ Cout,
                      int N, int K)
{
    const int lane = threadIdx.x & 31;
    const int wave = threadIdx.x >> 5;
    const int m0   = blockIdx.y * 32 + (wave >> 2) * 16;
    const int n0   = blockIdx.x * 256 + (wave & 3) * 64;
    const int l16  = lane & 15;
    const int half = lane >> 4;

    const bf16_t* Ap  = A + (size_t)(m0 + l16) * K + (half << 3);
    const bf16_t* Wp0 = W + (size_t)(n0 + l16) * K + (half << 3);
    const bf16_t* Wp1 = Wp0 + (size_t)16 * K;
    const bf16_t* Wp2 = Wp0 + (size_t)32 * K;
    const bf16_t* Wp3 = Wp0 + (size_t)48 * K;

    const v8f z = {0.f,0.f,0.f,0.f,0.f,0.f,0.f,0.f};
    v8f acc0 = z, acc1 = z, acc2 = z, acc3 = z;

#define LD_A(f, k) { *(uint4*)&(f).u[0] = *(const uint4*)(Ap + (k));            \
                     *(uint4*)&(f).u[4] = *(const uint4*)(Ap + (k) + 16); }
#define LD_B(f, P, k) { *(uint4*)&(f).u[0] = *(const uint4*)((P) + (k));        \
                        *(uint4*)&(f).u[4] = *(const uint4*)((P) + (k) + 16); }

    FragAB aC, b0C, b1C, b2C, b3C;
    LD_A(aC, 0);
    LD_B(b0C, Wp0, 0); LD_B(b1C, Wp1, 0); LD_B(b2C, Wp2, 0); LD_B(b3C, Wp3, 0);

    for (int k0 = 0; k0 < K; k0 += 32) {
        const int kn = (k0 + 32 < K) ? (k0 + 32) : 0;   // wrap: benign prefetch
        FragAB aN, b0N, b1N, b2N, b3N;
        LD_A(aN, kn);
        LD_B(b0N, Wp0, kn); LD_B(b1N, Wp1, kn);
        LD_B(b2N, Wp2, kn); LD_B(b3N, Wp3, kn);

        acc0 = __builtin_amdgcn_wmma_f32_16x16x32_bf16(false, aC.v, false, b0C.v,
                                                       (short)0, acc0, false, false);
        acc1 = __builtin_amdgcn_wmma_f32_16x16x32_bf16(false, aC.v, false, b1C.v,
                                                       (short)0, acc1, false, false);
        acc2 = __builtin_amdgcn_wmma_f32_16x16x32_bf16(false, aC.v, false, b2C.v,
                                                       (short)0, acc2, false, false);
        acc3 = __builtin_amdgcn_wmma_f32_16x16x32_bf16(false, aC.v, false, b3C.v,
                                                       (short)0, acc3, false, false);

        aC = aN; b0C = b0N; b1C = b1N; b2C = b2N; b3C = b3N;
    }
#undef LD_A
#undef LD_B

    // C/D layout: lane col = l16, VGPR r -> row m0 + r + half*8
    v8f accs[4] = {acc0, acc1, acc2, acc3};
    #pragma unroll
    for (int j = 0; j < 4; ++j) {
        const int col = n0 + (j << 4) + l16;
        const float bv = HAS_BIAS ? bias[col] : 0.0f;
        #pragma unroll
        for (int r = 0; r < 8; ++r) {
            const int row = m0 + r + (half << 3);
            const float val = accs[j][r] + bv;
            if (OUT_BF16)
                ((bf16_t*)Cout)[(size_t)row * N + col] = (bf16_t)val;
            else
                ((float*)Cout)[(size_t)row * N + col] = val;
        }
    }
}

// ---------------------------------------------------------------------------
// Prep kernels
// ---------------------------------------------------------------------------
__global__ void f32_to_bf16_kernel(const float* __restrict__ src,
                                   bf16_t* __restrict__ dst, long n)
{
    long i = (long)blockIdx.x * blockDim.x + threadIdx.x;
    const long stride = (long)gridDim.x * blockDim.x;
    for (; i < n; i += stride) dst[i] = (bf16_t)src[i];
}

// Wcat[n][0..255] = Wih[n][:], Wcat[n][256..383] = Whh[n][:]   (n < 512)
__global__ void build_wcat_kernel(const float* __restrict__ Wih,
                                  const float* __restrict__ Whh,
                                  bf16_t* __restrict__ dst)
{
    const int i = blockIdx.x * blockDim.x + threadIdx.x;
    if (i >= 512 * 384) return;
    const int n = i / 384, k = i % 384;
    const float v = (k < 256) ? Wih[n * 256 + k] : Whh[n * 128 + (k - 256)];
    dst[i] = (bf16_t)v;
}

__global__ void bias_combine_kernel(const float* __restrict__ a,
                                    const float* __restrict__ b,
                                    float* __restrict__ dst, int n)
{
    const int i = blockIdx.x * blockDim.x + threadIdx.x;
    if (i < n) dst[i] = a[i] + b[i];
}

// dec_ipt0 = tanh([1.1, 1/k] @ W_first^T + b_first); h0/c0 init; zero mask/lp
__global__ __launch_bounds__(256)
void init_kernel(const float* __restrict__ rk, const float* __restrict__ W_first,
                 const float* __restrict__ b_first,
                 const float* __restrict__ h0, const float* __restrict__ c0,
                 bf16_t* __restrict__ xh_f, bf16_t* __restrict__ xh_b,
                 float* __restrict__ c_f, float* __restrict__ c_b,
                 float* __restrict__ mask, float* __restrict__ out_lp)
{
    const int b = blockIdx.x, t = threadIdx.x;
    const float invk = 1.0f / rk[b];
    const float x0 = tanhf(1.1f * W_first[t * 2 + 0] + invk * W_first[t * 2 + 1]
                           + b_first[t]);
    const bf16_t xb = (bf16_t)x0;
    xh_f[(size_t)b * 384 + t] = xb;
    xh_b[(size_t)b * 384 + t] = xb;
    if (t < HID) {
        xh_f[(size_t)b * 384 + 256 + t] = (bf16_t)h0[(size_t)b * HID + t];
        xh_b[(size_t)b * 384 + 256 + t] =
            (bf16_t)h0[(size_t)BATCH * HID + (size_t)b * HID + t];
        c_f[(size_t)b * HID + t] = c0[(size_t)b * HID + t];
        c_b[(size_t)b * HID + t] = c0[(size_t)BATCH * HID + (size_t)b * HID + t];
    }
    if (t < SEQ) mask[(size_t)b * SEQ + t] = 0.0f;
    if (t == 0)  out_lp[b] = 0.0f;
}

// ---------------------------------------------------------------------------
// LSTM pointwise update (gates already contain Wx+Wh+bias). Gate order i,f,g,o.
// 256 threads/block: t<128 -> forward cell, t>=128 -> backward cell.
// ---------------------------------------------------------------------------
__global__ __launch_bounds__(256)
void lstm_update_kernel(const float* __restrict__ gates_f,
                        const float* __restrict__ gates_b,
                        float* __restrict__ c_f, float* __restrict__ c_b,
                        bf16_t* __restrict__ xh_f, bf16_t* __restrict__ xh_b,
                        bf16_t* __restrict__ qb)
{
    const int b = blockIdx.x;
    const int t = threadIdx.x;
    const int u = t & 127;
    const bool fwd = (t < 128);
    const float* g = (fwd ? gates_f : gates_b) + (size_t)b * 512;
    float* cp = (fwd ? c_f : c_b) + (size_t)b * HID + u;
    const float gi = g[u], gf = g[128 + u], gg = g[256 + u], go = g[384 + u];
    const float c2 = sigmoidf_(gf) * (*cp) + sigmoidf_(gi) * tanhf(gg);
    const float h2 = sigmoidf_(go) * tanhf(c2);
    *cp = c2;
    const bf16_t hb = (bf16_t)h2;
    qb[(size_t)b * DIM + t] = hb;                      // q = [hf, hb]
    (fwd ? xh_f : xh_b)[(size_t)b * 384 + 256 + u] = hb;
}

// ---------------------------------------------------------------------------
// Glimpse attention: one block per batch row. refg + enc tiles are TDM-staged
// to LDS, then:
//   u[s]   = sum_d vg[d]*tanh(refg[s,b,d] + qproj[b,d])
//   a      = softmax(mask? -inf : u)
//   g_l[d] = sum_s a[s]*enc[s,b,d]   -> bf16 out
// ---------------------------------------------------------------------------
__global__ __launch_bounds__(256)
void glimpse_kernel(const bf16_t* __restrict__ refg, const float* __restrict__ qproj,
                    const float* __restrict__ vg, const float* __restrict__ mask,
                    const bf16_t* __restrict__ encb, bf16_t* __restrict__ gl)
{
    const int b = blockIdx.x, t = threadIdx.x;
    __shared__ __align__(16) bf16_t tR[SEQ * DIM];
    __shared__ __align__(16) bf16_t tE[SEQ * DIM];
    __shared__ float qp[DIM], vw[DIM], su[64], sa[64];

    stage_tile(tR, refg, b, t);
    stage_tile(tE, encb, b, t);
    qp[t] = qproj[(size_t)b * DIM + t];
    vw[t] = vg[t];
    stage_wait(t);

    const int lane = t & 31, wave = t >> 5;
    for (int s = wave; s < SEQ; s += 8) {
        union { uint4 u4; bf16_t h[8]; } cv;
        cv.u4 = *reinterpret_cast<const uint4*>(tR + s * DIM + lane * 8);
        float sum = 0.0f;
        #pragma unroll
        for (int i = 0; i < 8; ++i) {
            const int d = lane * 8 + i;
            sum += vw[d] * tanhf((float)cv.h[i] + qp[d]);
        }
        #pragma unroll
        for (int off = 16; off > 0; off >>= 1) sum += __shfl_down(sum, off, 32);
        if (lane == 0) su[s] = sum;
    }
    __syncthreads();
    if (t < SEQ && mask[(size_t)b * SEQ + t] > 0.0f) su[t] = -1e30f;
    __syncthreads();

    if (t < 32) {
        const float l0 = (t < SEQ)      ? su[t]      : -1e30f;
        const float l1 = (t + 32 < SEQ) ? su[t + 32] : -1e30f;
        float m = fmaxf(l0, l1);
        #pragma unroll
        for (int off = 16; off > 0; off >>= 1) m = fmaxf(m, __shfl_xor(m, off, 32));
        const float e0 = (t < SEQ)      ? __expf(l0 - m) : 0.0f;
        const float e1 = (t + 32 < SEQ) ? __expf(l1 - m) : 0.0f;
        float se = e0 + e1;
        #pragma unroll
        for (int off = 16; off > 0; off >>= 1) se += __shfl_xor(se, off, 32);
        const float inv = 1.0f / se;
        if (t < SEQ)      sa[t]      = e0 * inv;
        if (t + 32 < SEQ) sa[t + 32] = e1 * inv;
    }
    __syncthreads();

    float acc = 0.0f;
    #pragma unroll 5
    for (int s = 0; s < SEQ; ++s)
        acc += sa[s] * (float)tE[s * DIM + t];
    gl[(size_t)b * DIM + t] = (bf16_t)acc;
}

// ---------------------------------------------------------------------------
// Pointer attention: refp tile TDM-staged; u[s] as above with (refp, v, pproj);
// argmax + lp = -log(sum exp(l - max)); mask update; next decoder input.
// ---------------------------------------------------------------------------
__global__ __launch_bounds__(256)
void pointer_kernel(const bf16_t* __restrict__ refp, const float* __restrict__ pproj,
                    const float* __restrict__ vvec, float* __restrict__ mask,
                    const bf16_t* __restrict__ encb,
                    bf16_t* __restrict__ xh_f, bf16_t* __restrict__ xh_b,
                    float* __restrict__ out_sel, float* __restrict__ out_lp, int step)
{
    const int b = blockIdx.x, t = threadIdx.x;
    __shared__ __align__(16) bf16_t tR[SEQ * DIM];
    __shared__ float qp[DIM], vw[DIM], su[64];
    __shared__ int ssel;

    stage_tile(tR, refp, b, t);
    qp[t] = pproj[(size_t)b * DIM + t];
    vw[t] = vvec[t];
    stage_wait(t);

    const int lane = t & 31, wave = t >> 5;
    for (int s = wave; s < SEQ; s += 8) {
        union { uint4 u4; bf16_t h[8]; } cv;
        cv.u4 = *reinterpret_cast<const uint4*>(tR + s * DIM + lane * 8);
        float sum = 0.0f;
        #pragma unroll
        for (int i = 0; i < 8; ++i) {
            const int d = lane * 8 + i;
            sum += vw[d] * tanhf((float)cv.h[i] + qp[d]);
        }
        #pragma unroll
        for (int off = 16; off > 0; off >>= 1) sum += __shfl_down(sum, off, 32);
        if (lane == 0) su[s] = sum;
    }
    __syncthreads();
    if (t < SEQ && mask[(size_t)b * SEQ + t] > 0.0f) su[t] = -1e30f;
    __syncthreads();

    if (t < 32) {
        const float l0 = (t < SEQ)      ? su[t]      : -3e30f;
        const float l1 = (t + 32 < SEQ) ? su[t + 32] : -3e30f;
        float bv; int bi;
        if (l0 >= l1) { bv = l0; bi = t; } else { bv = l1; bi = t + 32; }
        #pragma unroll
        for (int off = 16; off > 0; off >>= 1) {
            const float ov = __shfl_down(bv, off, 32);
            const int   oi = __shfl_down(bi, off, 32);
            if (ov > bv || (ov == bv && oi < bi)) { bv = ov; bi = oi; }
        }
        bv = __shfl(bv, 0, 32);
        bi = __shfl(bi, 0, 32);
        float e = 0.0f;
        if (t < SEQ)      e += __expf(l0 - bv);
        if (t + 32 < SEQ) e += __expf(l1 - bv);
        #pragma unroll
        for (int off = 16; off > 0; off >>= 1) e += __shfl_xor(e, off, 32);
        if (t == 0) {
            out_sel[(size_t)b * STEPS + step] = (float)bi;
            out_lp[b] += -__logf(e);
            mask[(size_t)b * SEQ + bi] += 1.0f;
            ssel = bi;
        }
    }
    __syncthreads();

    const int sel = ssel;
    const bf16_t xv = encb[((size_t)sel * BATCH + b) * DIM + t];
    xh_f[(size_t)b * 384 + t] = xv;
    xh_b[(size_t)b * 384 + t] = xv;
}

// ---------------------------------------------------------------------------
// Host driver
// ---------------------------------------------------------------------------
extern "C" void kernel_launch(void* const* d_in, const int* in_sizes, int n_in,
                              void* d_out, int out_size, void* d_ws, size_t ws_size,
                              hipStream_t stream)
{
    (void)in_sizes; (void)n_in; (void)out_size; (void)ws_size;

    const float* enc     = (const float*)d_in[0];
    const float* h0      = (const float*)d_in[1];
    const float* c0      = (const float*)d_in[2];
    const float* rk      = (const float*)d_in[3];
    const float* W_first = (const float*)d_in[4];
    const float* b_first = (const float*)d_in[5];
    const float* Wih_f   = (const float*)d_in[6];
    const float* Whh_f   = (const float*)d_in[7];
    const float* bih_f   = (const float*)d_in[8];
    const float* bhh_f   = (const float*)d_in[9];
    const float* Wih_b   = (const float*)d_in[10];
    const float* Whh_b   = (const float*)d_in[11];
    const float* bih_b   = (const float*)d_in[12];
    const float* bhh_b   = (const float*)d_in[13];
    const float* Wref_g  = (const float*)d_in[14];
    const float* Wq_g    = (const float*)d_in[15];
    const float* vg      = (const float*)d_in[16];
    const float* Wref    = (const float*)d_in[17];
    const float* Wq      = (const float*)d_in[18];
    const float* vv      = (const float*)d_in[19];

    float* out_sel = (float*)d_out;                       // [B, 49] as float
    float* out_lp  = out_sel + (size_t)BATCH * STEPS;     // [B]

    // ---- carve workspace ----
    char* p = (char*)d_ws;
    auto carve = [&](size_t bytes) -> char* {
        char* r = p; p += (bytes + 255) & ~(size_t)255; return r;
    };
    const size_t encN = (size_t)SEQ * BATCH * DIM;
    bf16_t* enc_bf   = (bf16_t*)carve(encN * 2);
    bf16_t* refg_bf  = (bf16_t*)carve(encN * 2);
    bf16_t* refp_bf  = (bf16_t*)carve(encN * 2);
    bf16_t* wcat_f   = (bf16_t*)carve((size_t)512 * 384 * 2);
    bf16_t* wcat_b   = (bf16_t*)carve((size_t)512 * 384 * 2);
    float*  bias_f   = (float*)carve(512 * 4);
    float*  bias_b   = (float*)carve(512 * 4);
    bf16_t* wqg_bf   = (bf16_t*)carve((size_t)DIM * DIM * 2);
    bf16_t* wq_bf    = (bf16_t*)carve((size_t)DIM * DIM * 2);
    bf16_t* wrefg_bf = (bf16_t*)carve((size_t)DIM * DIM * 2);
    bf16_t* wref_bf  = (bf16_t*)carve((size_t)DIM * DIM * 2);
    bf16_t* xh_f     = (bf16_t*)carve((size_t)BATCH * 384 * 2);
    bf16_t* xh_b     = (bf16_t*)carve((size_t)BATCH * 384 * 2);
    float*  gates_f  = (float*)carve((size_t)BATCH * 512 * 4);
    float*  gates_b  = (float*)carve((size_t)BATCH * 512 * 4);
    float*  c_f      = (float*)carve((size_t)BATCH * HID * 4);
    float*  c_b      = (float*)carve((size_t)BATCH * HID * 4);
    bf16_t* q_bf     = (bf16_t*)carve((size_t)BATCH * DIM * 2);
    float*  qproj    = (float*)carve((size_t)BATCH * DIM * 4);
    bf16_t* gl_bf    = (bf16_t*)carve((size_t)BATCH * DIM * 2);
    float*  pproj    = (float*)carve((size_t)BATCH * DIM * 4);
    float*  maskp    = (float*)carve((size_t)BATCH * SEQ * 4);

    const dim3 blk(256);

    // ---- prep: weight/activation conversion ----
    build_wcat_kernel<<<(512 * 384 + 255) / 256, blk, 0, stream>>>(Wih_f, Whh_f, wcat_f);
    build_wcat_kernel<<<(512 * 384 + 255) / 256, blk, 0, stream>>>(Wih_b, Whh_b, wcat_b);
    bias_combine_kernel<<<2, blk, 0, stream>>>(bih_f, bhh_f, bias_f, 512);
    bias_combine_kernel<<<2, blk, 0, stream>>>(bih_b, bhh_b, bias_b, 512);
    f32_to_bf16_kernel<<<256, blk, 0, stream>>>(Wq_g,   wqg_bf,   (long)DIM * DIM);
    f32_to_bf16_kernel<<<256, blk, 0, stream>>>(Wq,     wq_bf,    (long)DIM * DIM);
    f32_to_bf16_kernel<<<256, blk, 0, stream>>>(Wref_g, wrefg_bf, (long)DIM * DIM);
    f32_to_bf16_kernel<<<256, blk, 0, stream>>>(Wref,   wref_bf,  (long)DIM * DIM);
    f32_to_bf16_kernel<<<2048, blk, 0, stream>>>(enc, enc_bf, (long)encN);

    init_kernel<<<BATCH, blk, 0, stream>>>(rk, W_first, b_first, h0, c0,
                                           xh_f, xh_b, c_f, c_b, maskp, out_lp);

    // ---- precompute ref projections: [S*B, D] = enc_bf @ Wref*^T (bf16 out) ----
    gemm_bf16_kernel<true, false><<<dim3(1, (SEQ * BATCH) / 32), blk, 0, stream>>>(
        enc_bf, wrefg_bf, nullptr, refg_bf, DIM, DIM);
    gemm_bf16_kernel<true, false><<<dim3(1, (SEQ * BATCH) / 32), blk, 0, stream>>>(
        enc_bf, wref_bf, nullptr, refp_bf, DIM, DIM);

    // ---- 49 sequential decode steps ----
    for (int step = 0; step < STEPS; ++step) {
        // LSTM gates: [B,512] = xh[B,384] @ Wcat^T + bias
        gemm_bf16_kernel<false, true><<<dim3(2, BATCH / 32), blk, 0, stream>>>(
            xh_f, wcat_f, bias_f, gates_f, 512, 384);
        gemm_bf16_kernel<false, true><<<dim3(2, BATCH / 32), blk, 0, stream>>>(
            xh_b, wcat_b, bias_b, gates_b, 512, 384);

        lstm_update_kernel<<<BATCH, blk, 0, stream>>>(gates_f, gates_b, c_f, c_b,
                                                      xh_f, xh_b, q_bf);

        // qproj = q @ Wq_g^T
        gemm_bf16_kernel<false, false><<<dim3(1, BATCH / 32), blk, 0, stream>>>(
            q_bf, wqg_bf, nullptr, qproj, DIM, DIM);

        glimpse_kernel<<<BATCH, blk, 0, stream>>>(refg_bf, qproj, vg, maskp,
                                                  enc_bf, gl_bf);

        // pproj = g_l @ Wq^T
        gemm_bf16_kernel<false, false><<<dim3(1, BATCH / 32), blk, 0, stream>>>(
            gl_bf, wq_bf, nullptr, pproj, DIM, DIM);

        pointer_kernel<<<BATCH, blk, 0, stream>>>(refp_bf, pproj, vv, maskp,
                                                  enc_bf, xh_f, xh_b,
                                                  out_sel, out_lp, step);
    }
}
